// GAT_27084063769013
// MI455X (gfx1250) — matrix operand order repaired and checked by reading, not compiled, not run on hardware
//
#include <hip/hip_runtime.h>

// ---------------------------------------------------------------------------
// Types for CDNA5 WMMA (wave32): bf16 A/B as 16-element ext vectors, f32 C/D.
// ---------------------------------------------------------------------------
typedef __attribute__((ext_vector_type(16))) __bf16         v16bf;
typedef __attribute__((ext_vector_type(16))) unsigned short v16u;
typedef __attribute__((ext_vector_type(8)))  unsigned short v8u;
typedef __attribute__((ext_vector_type(8)))  float          v8f;
typedef __attribute__((ext_vector_type(4)))  float          v4f;

#define NEG_SLOPE 0.2f

__device__ __forceinline__ unsigned short f32_to_bf16(float f) {
  // round-to-nearest-even to bf16
  unsigned int u = __float_as_uint(f);
  unsigned int r = (u + 0x7FFFu + ((u >> 16) & 1u)) >> 16;
  return (unsigned short)r;
}

// Float atomic max via the signed/unsigned int monotonicity trick
// (lowers to native GLOBAL_ATOMIC_MAX_I32 / MIN_U32 on gfx1250).
__device__ __forceinline__ void atomic_max_f32(float* addr, float val) {
  if (val >= 0.0f) atomicMax((int*)addr, __float_as_int(val));
  else             atomicMin((unsigned int*)addr, __float_as_uint(val));
}

// ---------------------------------------------------------------------------
// Utility kernels
// ---------------------------------------------------------------------------
__global__ void k_fill(float* p, float v, int n) {
  int i = blockIdx.x * blockDim.x + threadIdx.x;
  if (i < n) p[i] = v;
}

__global__ void k_fill_bias(float* p, const float* __restrict__ bias, int cols, int n) {
  int i = blockIdx.x * blockDim.x + threadIdx.x;
  if (i < n) p[i] = bias[i % cols];
}

// Convert W[K,NC] (f32 row-major) -> Wt[NC,K] (bf16, K-major). K-major storage
// makes each WMMA lane's two 8-element K runs contiguous (2 x b128 loads).
__global__ void k_cvt_bf16_t(const float* __restrict__ in, unsigned short* __restrict__ out,
                             int K, int NC) {
  int i = blockIdx.x * blockDim.x + threadIdx.x;
  if (i >= K * NC) return;
  int k = i / NC, n = i - k * NC;
  out[(size_t)n * K + k] = f32_to_bf16(in[i]);
}

// ---------------------------------------------------------------------------
// WMMA GEMM: C[M,NC] = A[M,K] (f32, cvt to bf16 on the fly) x Bt[NC,K] (bf16).
// Each wave32 computes a 16 x (16*NT) strip: A converted once per k-step and
// reused across NT column tiles -> NT back-to-back v_wmma per k-step.
//
// A-layout (16x32 bf16): lane half hf holds K = hf*8+e (e<8) / 16+hf*8+(e-8).
// B-layout (32x16 bf16): lane half hf holds K = hf*8+e (e<8) / 16+hf*8+(e-8), col = lane&15.
// D-layout (16x16 f32):  row = hf*8 + i, col = lane&15.
// ---------------------------------------------------------------------------
template <int K, int NC, int NT>
__global__ __launch_bounds__(256) void k_wmma_gemm(const float* __restrict__ A,
                                                   const unsigned short* __restrict__ Bt,
                                                   float* __restrict__ C,
                                                   int totalWaves) {
  int wave = (int)((blockIdx.x * blockDim.x + threadIdx.x) >> 5);
  int lane = (int)(threadIdx.x & 31);
  if (wave >= totalWaves) return;  // wave-uniform: EXEC all-ones at every WMMA

  const int groupsPerRow = NC / (16 * NT);
  int tm = wave / groupsPerRow;
  int tg = wave - tm * groupsPerRow;
  int hf  = lane >> 4;
  int l15 = lane & 15;

  const float* arow = A + (size_t)(tm * 16 + l15) * K;
  int col0 = tg * (16 * NT) + l15;

  v8f acc[NT];
#pragma unroll
  for (int j = 0; j < NT; ++j) acc[j] = (v8f){};

#pragma unroll
  for (int kb = 0; kb < K; kb += 32) {
    // ---- A: two contiguous 8-float runs -> 16 bf16 (vector f32 loads) ----
    const v4f a0 = *(const v4f*)(arow + kb + hf * 8);
    const v4f a1 = *(const v4f*)(arow + kb + hf * 8 + 4);
    const v4f a2 = *(const v4f*)(arow + kb + 16 + hf * 8);
    const v4f a3 = *(const v4f*)(arow + kb + 16 + hf * 8 + 4);
    v16u au;
#pragma unroll
    for (int e = 0; e < 4; ++e) {
      au[e]      = f32_to_bf16(a0[e]);
      au[4 + e]  = f32_to_bf16(a1[e]);
      au[8 + e]  = f32_to_bf16(a2[e]);
      au[12 + e] = f32_to_bf16(a3[e]);
    }
    v16bf a = __builtin_bit_cast(v16bf, au);

    // ---- B: K-major -> two 8-bf16 runs per lane, K=hf*8 and 16+hf*8 (2 x 16B loads) ----
#pragma unroll
    for (int j = 0; j < NT; ++j) {
      const unsigned short* bp = Bt + (size_t)(col0 + j * 16) * K + kb + hf * 8;
      v8u b0 = *(const v8u*)(bp);
      v8u b1 = *(const v8u*)(bp + 16);
      v16u bu = __builtin_shufflevector(b0, b1, 0, 1, 2, 3, 4, 5, 6, 7,
                                        8, 9, 10, 11, 12, 13, 14, 15);
      v16bf b = __builtin_bit_cast(v16bf, bu);
      acc[j] = __builtin_amdgcn_wmma_f32_16x16x32_bf16(false, a, false, b,
                                                       (short)0, acc[j], false, false);
    }
  }

#pragma unroll
  for (int j = 0; j < NT; ++j)
#pragma unroll
    for (int i = 0; i < 8; ++i)
      C[(size_t)(tm * 16 + hf * 8 + i) * NC + col0 + j * 16] = acc[j][i];
}

// ---------------------------------------------------------------------------
// Per-node attention scores: as[n,h] = sum_c h[n,h,c]*att_s[h,c]; same for ad.
// ---------------------------------------------------------------------------
template <int H, int C>
__global__ void k_att_scores(const float* __restrict__ hmat,
                             const float* __restrict__ att_s,
                             const float* __restrict__ att_d,
                             float* __restrict__ as, float* __restrict__ ad, int N) {
  int t = blockIdx.x * blockDim.x + threadIdx.x;
  if (t >= N * H) return;
  int n = t / H, h = t - n * H;
  const float* row = hmat + (size_t)(n * H + h) * C;
  float s = 0.f, d = 0.f;
#pragma unroll 4
  for (int c = 0; c < C; ++c) {
    float v = row[c];
    s += v * att_s[h * C + c];
    d += v * att_d[h * C + c];
  }
  as[t] = s;
  ad[t] = d;
}

// ---------------------------------------------------------------------------
// Edge passes. Edge e < E comes from edge_index; e >= E is the self-loop e-E.
// ---------------------------------------------------------------------------
__device__ __forceinline__ void edge_sd(const int* __restrict__ ei, int E, int e,
                                        int& s, int& d) {
  if (e < E) { s = ei[e]; d = ei[E + e]; }
  else       { s = e - E; d = e - E; }
}

template <int H>
__global__ void k_edge_max(const int* __restrict__ ei, const float* __restrict__ as,
                           const float* __restrict__ ad, float* __restrict__ m,
                           int E, int N) {
  int t = blockIdx.x * blockDim.x + threadIdx.x;
  if (t >= (E + N) * H) return;
  int e = t / H, h = t - e * H;
  int s, d; edge_sd(ei, E, e, s, d);
  float v = as[s * H + h] + ad[d * H + h];
  v = v > 0.f ? v : NEG_SLOPE * v;
  atomic_max_f32(&m[d * H + h], v);
}

template <int H>
__global__ void k_edge_expsum(const int* __restrict__ ei, const float* __restrict__ as,
                              const float* __restrict__ ad, const float* __restrict__ m,
                              float* __restrict__ ssum, int E, int N) {
  int t = blockIdx.x * blockDim.x + threadIdx.x;
  if (t >= (E + N) * H) return;
  int e = t / H, h = t - e * H;
  int s, d; edge_sd(ei, E, e, s, d);
  float v = as[s * H + h] + ad[d * H + h];
  v = v > 0.f ? v : NEG_SLOPE * v;
  atomicAdd(&ssum[d * H + h], __expf(v - m[d * H + h]));
}

// One thread per (edge, channel): wave-coalesced gather of h[src] and
// wave-coalesced atomicAdd scatter into o[dst]. alpha recompute is cheap and
// its operands (as/ad/m/ssum) are L2-resident.
template <int H, int C>
__global__ void k_edge_agg(const int* __restrict__ ei, const float* __restrict__ as,
                           const float* __restrict__ ad, const float* __restrict__ m,
                           const float* __restrict__ ssum, const float* __restrict__ hmat,
                           float* __restrict__ o, int E, int N) {
  const int HC = H * C;
  long long t = (long long)blockIdx.x * blockDim.x + threadIdx.x;
  if (t >= (long long)(E + N) * HC) return;
  int e   = (int)(t / HC);            // HC is a power of two -> shift
  int rem = (int)(t - (long long)e * HC);
  int h   = rem / C;                  // power of two -> shift
  int s, d; edge_sd(ei, E, e, s, d);
  int dh = d * H + h;
  float v = as[s * H + h] + ad[dh];
  v = v > 0.f ? v : NEG_SLOPE * v;
  float alpha = __expf(v - m[dh]) / (ssum[dh] + 1e-16f);
  atomicAdd(&o[(size_t)d * HC + rem], hmat[(size_t)s * HC + rem] * alpha);
}

__global__ void k_bias_relu(float* __restrict__ o, const float* __restrict__ bias,
                            int cols, int n) {
  int i = blockIdx.x * blockDim.x + threadIdx.x;
  if (i < n) o[i] = fmaxf(o[i] + bias[i % cols], 0.f);
}

// ---------------------------------------------------------------------------
// Host-side orchestration
// ---------------------------------------------------------------------------
static inline unsigned gdim(long long n) { return (unsigned)((n + 255) / 256); }

extern "C" void kernel_launch(void* const* d_in, const int* in_sizes, int n_in,
                              void* d_out, int out_size, void* d_ws, size_t ws_size,
                              hipStream_t stream) {
  const float* x      = (const float*)d_in[0];
  const int*   ei     = (const int*)  d_in[1];
  const float* W1     = (const float*)d_in[2];
  const float* att_s1 = (const float*)d_in[3];
  const float* att_d1 = (const float*)d_in[4];
  const float* b1     = (const float*)d_in[5];
  const float* W2     = (const float*)d_in[6];
  const float* att_s2 = (const float*)d_in[7];
  const float* att_d2 = (const float*)d_in[8];
  const float* b2     = (const float*)d_in[9];
  float* out = (float*)d_out;

  const int IN = 128, H1 = 8, C1 = 32, HC1 = 256, OUT = 64;
  const int N = in_sizes[0] / IN;      // 50000 (divisible by 16)
  const int E = in_sizes[1] / 2;       // 800000
  (void)ws_size; (void)n_in; (void)out_size;

  // Workspace carve-up (~123 MB total)
  char* base = (char*)d_ws;
  size_t off = 0;
  auto alloc = [&](size_t bytes) -> void* {
    void* p = base + off;
    off += bytes; off = (off + 255) & ~(size_t)255;
    return p;
  };
  float* h1  = (float*)alloc((size_t)N * HC1 * 4);   // x @ W1
  float* o1  = (float*)alloc((size_t)N * HC1 * 4);   // layer-1 aggregated + relu
  float* h2  = (float*)alloc((size_t)N * OUT * 4);   // o1 @ W2
  float* as1 = (float*)alloc((size_t)N * H1 * 4);
  float* ad1 = (float*)alloc((size_t)N * H1 * 4);
  float* m1  = (float*)alloc((size_t)N * H1 * 4);
  float* s1  = (float*)alloc((size_t)N * H1 * 4);
  float* as2 = (float*)alloc((size_t)N * 4);
  float* ad2 = (float*)alloc((size_t)N * 4);
  float* m2  = (float*)alloc((size_t)N * 4);
  float* s2  = (float*)alloc((size_t)N * 4);
  unsigned short* w1t = (unsigned short*)alloc((size_t)IN * HC1 * 2);   // [HC1, IN]
  unsigned short* w2t = (unsigned short*)alloc((size_t)HC1 * OUT * 2);  // [OUT, HC1]

  // Weight conversion to bf16, transposed to K-major for contiguous WMMA loads
  k_cvt_bf16_t<<<gdim(IN * HC1), 256, 0, stream>>>(W1, w1t, IN, HC1);
  k_cvt_bf16_t<<<gdim(HC1 * OUT), 256, 0, stream>>>(W2, w2t, HC1, OUT);

  // ---- Layer 1: h1 = x @ W1 ----
  {
    int waves = (N / 16) * (HC1 / 64);   // 4 column tiles per wave
    k_wmma_gemm<IN, HC1, 4><<<gdim((long long)waves * 32), 256, 0, stream>>>(x, w1t, h1, waves);
  }
  k_att_scores<H1, C1><<<gdim((long long)N * H1), 256, 0, stream>>>(h1, att_s1, att_d1, as1, ad1, N);

  k_fill<<<gdim((long long)N * H1), 256, 0, stream>>>(m1, -1e30f, N * H1);
  k_fill<<<gdim((long long)N * H1), 256, 0, stream>>>(s1, 0.f, N * H1);
  k_fill<<<gdim((long long)N * HC1), 256, 0, stream>>>(o1, 0.f, N * HC1);

  long long totEH1 = (long long)(E + N) * H1;
  k_edge_max<H1>   <<<gdim(totEH1), 256, 0, stream>>>(ei, as1, ad1, m1, E, N);
  k_edge_expsum<H1><<<gdim(totEH1), 256, 0, stream>>>(ei, as1, ad1, m1, s1, E, N);
  k_edge_agg<H1, C1><<<gdim((long long)(E + N) * HC1), 256, 0, stream>>>(
      ei, as1, ad1, m1, s1, h1, o1, E, N);

  k_bias_relu<<<gdim((long long)N * HC1), 256, 0, stream>>>(o1, b1, HC1, N * HC1);

  // ---- Layer 2: h2 = o1 @ W2 ----
  {
    int waves = (N / 16) * (OUT / 64);   // 4 column tiles per wave = full row
    k_wmma_gemm<HC1, OUT, 4><<<gdim((long long)waves * 32), 256, 0, stream>>>(o1, w2t, h2, waves);
  }
  k_att_scores<1, OUT><<<gdim((long long)N), 256, 0, stream>>>(h2, att_s2, att_d2, as2, ad2, N);

  k_fill<<<gdim((long long)N), 256, 0, stream>>>(m2, -1e30f, N);
  k_fill<<<gdim((long long)N), 256, 0, stream>>>(s2, 0.f, N);
  // seed output with bias b2; edge aggregation adds the segment sums on top
  k_fill_bias<<<gdim((long long)N * OUT), 256, 0, stream>>>(out, b2, OUT, N * OUT);

  long long totEH2 = (long long)(E + N);
  k_edge_max<1>   <<<gdim(totEH2), 256, 0, stream>>>(ei, as2, ad2, m2, E, N);
  k_edge_expsum<1><<<gdim(totEH2), 256, 0, stream>>>(ei, as2, ad2, m2, s2, E, N);
  k_edge_agg<1, OUT><<<gdim((long long)(E + N) * OUT), 256, 0, stream>>>(
      ei, as2, ad2, m2, s2, h2, out, E, N);
}
